// Attention_60962765799812
// MI455X (gfx1250) — compile-verified, matrix operand order checked by reading, not run on hardware
//
#include <hip/hip_runtime.h>
#include <hip/hip_bf16.h>

// ---------------------------------------------------------------- types
typedef __bf16 bf16_t;
typedef __bf16 v16bf __attribute__((ext_vector_type(16)));
typedef __bf16 v8bf  __attribute__((ext_vector_type(8)));
typedef float  v8f   __attribute__((ext_vector_type(8)));

#define S_LEN 2048
#define HID   2048
#define NH    16
#define NKV   4
#define HD    128
#define WIN   1024
#define QKV_N 3072   // 2048 q + 512 k + 512 v

#define TILE_LD 40   // padded LDS row stride (elements): 32 data + 8 pad, 16B aligned

// ---------------------------------------------------------------- helpers
__device__ __forceinline__ v8f wmma_bf16(v16bf a, v16bf b, v8f c) {
  // D = A(16x32) * B(32x16) + C, f32 accumulate
  return __builtin_amdgcn_wmma_f32_16x16x32_bf16(
      /*neg_a=*/false, a, /*neg_b=*/false, b,
      /*c_mod=*/(short)0, c, /*reuse_a=*/false, /*reuse_b=*/false);
}

// Async DMA: global (16B per lane) -> LDS, tracked by ASYNCcnt (no VGPR data).
__device__ __forceinline__ void async_copy_b128(unsigned lds_addr, const bf16_t* gaddr) {
  asm volatile("global_load_async_to_lds_b128 %0, %1, off"
               :: "v"(lds_addr), "v"(gaddr) : "memory");
}
__device__ __forceinline__ void wait_async_le6() {
  asm volatile("s_wait_asynccnt 6" ::: "memory");
}
__device__ __forceinline__ void wait_async_0() {
  asm volatile("s_wait_asynccnt 0" ::: "memory");
}

// Load one 16-bit WMMA A/B fragment from a row-major [rows x ld] bf16 buffer (global).
// Per-lane layout (CDNA5): lanes 0-15: elems0-7 = K[k0..k0+7], elems8-15 = K[k0+16..k0+23]
//                          lanes16-31: same +8.  row already includes (lane&15).
__device__ __forceinline__ v16bf load_frag(const bf16_t* __restrict__ p,
                                           int ld, int row, int k0, int lane) {
  const bf16_t* q = p + row * ld + k0 + ((lane >> 4) << 3);
  v8bf lo = *(const v8bf*)q;
  v8bf hi = *(const v8bf*)(q + 16);
  v16bf f;
#pragma unroll
  for (int i = 0; i < 8; ++i) { f[i] = lo[i]; f[i + 8] = hi[i]; }
  return f;
}

// Same fragment gather out of an LDS tile (k-chunk starts at column 0).
__device__ __forceinline__ v16bf lds_frag(const bf16_t* t, int row, int lane) {
  const bf16_t* p = t + row * TILE_LD + ((lane >> 4) << 3);
  v8bf lo = *(const v8bf*)p;        // ds_load_b128
  v8bf hi = *(const v8bf*)(p + 16); // ds_load_b128
  v16bf f;
#pragma unroll
  for (int i = 0; i < 8; ++i) { f[i] = lo[i]; f[i + 8] = hi[i]; }
  return f;
}

// ---------------------------------------------------------------- convert
__global__ void f32_to_bf16_k(const float* __restrict__ in, bf16_t* __restrict__ out, int n) {
  int i = blockIdx.x * blockDim.x + threadIdx.x;
  if (i < n) out[i] = (bf16_t)in[i];
}

// ---------------------------------------------------------------- GEMM  C[M,N] = A[M,K] * B[N,K]^T  (bf16 in, f32 out)
// 256 threads = 8 wave32 (2x4); block tile 128x256, wave tile 64x64, k-step 32.
// A/B k-slices staged to LDS with async b128 DMA, double-buffered so the DMA of
// step t+1 overlaps the 16 WMMAs of step t.
__global__ __launch_bounds__(256)
void gemm_bf16_nt(const bf16_t* __restrict__ A, const bf16_t* __restrict__ B,
                  float* __restrict__ C, int M, int N, int K) {
  __shared__ bf16_t sA[2][128 * TILE_LD];   // 20 KB
  __shared__ bf16_t sB[2][256 * TILE_LD];   // 40 KB

  const int tid  = threadIdx.x;
  const int lane = tid & 31;
  const int wave = tid >> 5;
  const int wy = wave >> 2, wx = wave & 3;
  const int mblk = blockIdx.y * 128;
  const int nblk = blockIdx.x * 256;
  const int rl  = lane & 15;
  const int hi8 = (lane >> 4) << 3;
  (void)M;

  v8f acc[4][4];
#pragma unroll
  for (int i = 0; i < 4; ++i)
#pragma unroll
    for (int j = 0; j < 4; ++j) acc[i][j] = (v8f)0.f;

  // stage one 32-wide k-slice into LDS buffer `buf` (6 async ops per thread)
  auto stage = [&](int buf, int k0) {
#pragma unroll
    for (int s = 0; s < 2; ++s) {           // A: 128 rows x 4 chunks = 512 tasks
      int task = tid * 2 + s;
      int row = task >> 2, ch = (task & 3) << 3;
      async_copy_b128((unsigned)(size_t)&sA[buf][row * TILE_LD + ch],
                      A + (size_t)(mblk + row) * K + k0 + ch);
    }
#pragma unroll
    for (int s = 0; s < 4; ++s) {           // B: 256 rows x 4 chunks = 1024 tasks
      int task = tid * 4 + s;
      int row = task >> 2, ch = (task & 3) << 3;
      async_copy_b128((unsigned)(size_t)&sB[buf][row * TILE_LD + ch],
                      B + (size_t)(nblk + row) * K + k0 + ch);
    }
  };

  const int nk = K >> 5;
  stage(0, 0);
  for (int t = 0; t < nk; ++t) {
    const int cur = t & 1;
    if (t + 1 < nk) {
      stage(cur ^ 1, (t + 1) << 5);
      wait_async_le6();     // current slice landed; next slice still in flight
    } else {
      wait_async_0();
    }
    __syncthreads();

    v16bf af[4];
#pragma unroll
    for (int i = 0; i < 4; ++i)
      af[i] = lds_frag(sA[cur], wy * 64 + 16 * i + rl, lane);
#pragma unroll
    for (int j = 0; j < 4; ++j) {
      v16bf bfj = lds_frag(sB[cur], wx * 64 + 16 * j + rl, lane);
#pragma unroll
      for (int i = 0; i < 4; ++i)
        acc[i][j] = wmma_bf16(af[i], bfj, acc[i][j]);
    }
    __syncthreads();        // everyone done reading `cur` before it is re-staged
  }

  // D layout: lane = column n (lane&15); vgpr r holds row m = (lane>>4)*8 + r
#pragma unroll
  for (int i = 0; i < 4; ++i)
#pragma unroll
    for (int j = 0; j < 4; ++j) {
      int n = nblk + wx * 64 + 16 * j + rl;
      int m0 = mblk + wy * 64 + 16 * i + hi8;
#pragma unroll
      for (int r = 0; r < 8; ++r)
        C[(m0 + r) * N + n] = acc[i][j][r];
    }
}

// ---------------------------------------------------------------- RoPE + head-major bf16 pack
__global__ void rope_qk_k(const float* __restrict__ qkv,
                          bf16_t* __restrict__ Qb, bf16_t* __restrict__ Kb) {
  int tid = blockIdx.x * blockDim.x + threadIdx.x;
  const int nq = S_LEN * NH * 64;
  const int nk = S_LEN * NKV * 64;
  const float kLn1e4_64 = 0.14391156831212788f;  // ln(10000)/64
  if (tid < nq) {
    int j = tid & 63;
    int h = (tid >> 6) & (NH - 1);
    int s = tid >> 10;
    const float* base = qkv + (size_t)s * QKV_N + h * HD;
    float x1 = base[j], x2 = base[j + 64];
    float ang = (float)s * __expf(-(float)j * kLn1e4_64);
    float sn, cs; __sincosf(ang, &sn, &cs);
    bf16_t* out = Qb + ((size_t)h * S_LEN + s) * HD;
    out[j]      = (bf16_t)(x1 * cs - x2 * sn);
    out[j + 64] = (bf16_t)(x2 * cs + x1 * sn);
  } else if (tid < nq + nk) {
    int t = tid - nq;
    int j = t & 63;
    int h = (t >> 6) & (NKV - 1);
    int s = t >> 8;
    const float* base = qkv + (size_t)s * QKV_N + HID + h * HD;
    float x1 = base[j], x2 = base[j + 64];
    float ang = (float)s * __expf(-(float)j * kLn1e4_64);
    float sn, cs; __sincosf(ang, &sn, &cs);
    bf16_t* out = Kb + ((size_t)h * S_LEN + s) * HD;
    out[j]      = (bf16_t)(x1 * cs - x2 * sn);
    out[j + 64] = (bf16_t)(x2 * cs + x1 * sn);
  }
}

// V -> Vt [h][d][s] bf16 (transposed so Vt tiles load as WMMA A-operands)
__global__ void pack_vt_k(const float* __restrict__ qkv, bf16_t* __restrict__ Vt) {
  int tid = blockIdx.x * blockDim.x + threadIdx.x;
  if (tid >= NKV * HD * S_LEN) return;
  int s = tid & (S_LEN - 1);
  int d = (tid >> 11) & (HD - 1);
  int h = tid >> 18;
  Vt[tid] = (bf16_t)qkv[(size_t)s * QKV_N + HID + NKV * HD + h * HD + d];
}

// ---------------------------------------------------------------- flash attention (sliding window, GQA)
// One wave per (head, 16-query tile).  St = K*Q^T so D-layout lane = query column;
// P registers then match the B-operand layout of Ot = Vt*P^T exactly (no shuffles).
__global__ __launch_bounds__(256)
void attn_fa_k(const bf16_t* __restrict__ Qb, const bf16_t* __restrict__ Kb,
               const bf16_t* __restrict__ Vt, bf16_t* __restrict__ Ob) {
  const int lane = threadIdx.x & 31;
  const int wave = threadIdx.x >> 5;
  const int g   = blockIdx.x * 8 + wave;   // 0 .. NH*(S/16)-1
  const int h   = g >> 7;                  // / (S_LEN/16)
  const int q0  = (g & 127) << 4;
  const int kvh = h >> 2;                  // NREP = 4
  const int rl  = lane & 15;
  const int hi8 = (lane >> 4) << 3;
  const int q_mine = q0 + rl;

  const bf16_t* Qh = Qb + (size_t)h   * S_LEN * HD;
  const bf16_t* Kh = Kb + (size_t)kvh * S_LEN * HD;
  const bf16_t* Vh = Vt + (size_t)kvh * HD * S_LEN;

  v16bf qf[4];
#pragma unroll
  for (int c = 0; c < 4; ++c) qf[c] = load_frag(Qh, HD, q0 + rl, 32 * c, lane);

  v8f ot[8];
#pragma unroll
  for (int db = 0; db < 8; ++db) ot[db] = (v8f)0.f;
  float m_run = -1e30f, l_run = 0.f;

  int kstart = q0 - (WIN - 1); if (kstart < 0) kstart = 0; kstart &= ~31;
  const int kend = q0 + 15;
  const float scale = 0.08838834764831845f;  // 1/sqrt(128)

  for (int k0 = kstart; k0 <= kend; k0 += 32) {
    v8f s0 = (v8f)0.f, s1 = (v8f)0.f;
#pragma unroll
    for (int c = 0; c < 4; ++c)
      s0 = wmma_bf16(load_frag(Kh, HD, k0 + rl, 32 * c, lane), qf[c], s0);
#pragma unroll
    for (int c = 0; c < 4; ++c)
      s1 = wmma_bf16(load_frag(Kh, HD, k0 + 16 + rl, 32 * c, lane), qf[c], s1);

    float sv[16];
    float cmax = -1e30f;
#pragma unroll
    for (int r = 0; r < 8; ++r) {
      int key0 = k0 + hi8 + r;
      int key1 = key0 + 16;
      float a0 = s0[r] * scale, a1 = s1[r] * scale;
      a0 = ((key0 <= q_mine) && (q_mine - key0 < WIN)) ? a0 : -1e30f;
      a1 = ((key1 <= q_mine) && (q_mine - key1 < WIN)) ? a1 : -1e30f;
      sv[r] = a0; sv[r + 8] = a1;
      cmax = fmaxf(cmax, fmaxf(a0, a1));
    }
    cmax = fmaxf(cmax, __shfl_xor(cmax, 16, 32));
    float m_new = fmaxf(m_run, cmax);
    float resc  = __expf(m_run - m_new);

    float psum = 0.f;
    v16bf pb;
#pragma unroll
    for (int r = 0; r < 16; ++r) {
      float p = __expf(sv[r] - m_new);
      psum += p;
      pb[r] = (bf16_t)p;
    }
    psum += __shfl_xor(psum, 16, 32);
    l_run = l_run * resc + psum;
    m_run = m_new;

#pragma unroll
    for (int db = 0; db < 8; ++db) {
#pragma unroll
      for (int r = 0; r < 8; ++r) ot[db][r] *= resc;
    }
#pragma unroll
    for (int db = 0; db < 8; ++db) {
      v16bf vf = load_frag(Vh, S_LEN, db * 16 + rl, k0, lane);
      ot[db] = wmma_bf16(vf, pb, ot[db]);  // Ot[d][q] accumulate
    }
  }

  float inv_l = 1.f / l_run;
#pragma unroll
  for (int db = 0; db < 8; ++db) {
    v8bf o;
#pragma unroll
    for (int r = 0; r < 8; ++r) o[r] = (bf16_t)(ot[db][r] * inv_l);
    bf16_t* dst = Ob + (size_t)q_mine * (NH * HD) + h * HD + db * 16 + hi8;
    *(v8bf*)dst = o;
  }
}

// ---------------------------------------------------------------- launch
extern "C" void kernel_launch(void* const* d_in, const int* in_sizes, int n_in,
                              void* d_out, int out_size, void* d_ws, size_t ws_size,
                              hipStream_t stream) {
  (void)in_sizes; (void)n_in; (void)out_size; (void)ws_size;
  const float* x  = (const float*)d_in[0];
  // d_in[1] = attention_mask (unused; mask computed analytically)
  const float* wq = (const float*)d_in[2];
  const float* wk = (const float*)d_in[3];
  const float* wv = (const float*)d_in[4];
  const float* wo = (const float*)d_in[5];

  char* ws = (char*)d_ws;
  bf16_t* xb    = (bf16_t*)ws;  ws += (size_t)S_LEN * HID * 2;
  bf16_t* wqkvb = (bf16_t*)ws;  ws += (size_t)QKV_N * HID * 2;
  bf16_t* wob   = (bf16_t*)ws;  ws += (size_t)HID * HID * 2;
  float*  qkv   = (float*) ws;  ws += (size_t)S_LEN * QKV_N * 4;
  bf16_t* Qb    = (bf16_t*)ws;  ws += (size_t)NH * S_LEN * HD * 2;
  bf16_t* Kb    = (bf16_t*)ws;  ws += (size_t)NKV * S_LEN * HD * 2;
  bf16_t* Vt    = (bf16_t*)ws;  ws += (size_t)NKV * HD * S_LEN * 2;
  bf16_t* attnb = (bf16_t*)ws;  ws += (size_t)S_LEN * NH * HD * 2;

  auto conv = [&](const float* src, bf16_t* dst, int n) {
    f32_to_bf16_k<<<(n + 255) / 256, 256, 0, stream>>>(src, dst, n);
  };
  conv(x,  xb,    S_LEN * HID);
  conv(wq, wqkvb,                          NH  * HD * HID);
  conv(wk, wqkvb + (size_t)NH * HD * HID,  NKV * HD * HID);
  conv(wv, wqkvb + (size_t)(NH + NKV) * HD * HID, NKV * HD * HID);
  conv(wo, wob,   HID * HID);

  // QKV projection: [2048,3072] = xb[2048,2048] * wqkvb[3072,2048]^T
  dim3 g1(QKV_N / 256, S_LEN / 128);
  gemm_bf16_nt<<<g1, 256, 0, stream>>>(xb, wqkvb, qkv, S_LEN, QKV_N, HID);

  int nrope = S_LEN * NH * 64 + S_LEN * NKV * 64;
  rope_qk_k<<<(nrope + 255) / 256, 256, 0, stream>>>(qkv, Qb, Kb);
  int nv = NKV * HD * S_LEN;
  pack_vt_k<<<(nv + 255) / 256, 256, 0, stream>>>(qkv, Vt);

  // flash attention: NH * (S/16) waves, 8 waves / block
  attn_fa_k<<<(NH * (S_LEN / 16)) / 8, 256, 0, stream>>>(Qb, Kb, Vt, attnb);

  // output projection: d_out[2048,2048] = attnb * wob^T
  dim3 g2(HID / 256, S_LEN / 128);
  gemm_bf16_nt<<<g2, 256, 0, stream>>>(attnb, wob, (float*)d_out, S_LEN, HID, HID);
}
